// PFE_50629074485701
// MI455X (gfx1250) — matrix operand order repaired and checked by reading, not compiled, not run on hardware
//
#include <hip/hip_runtime.h>
#include <hip/hip_bf16.h>

#define EMBED 384

typedef __attribute__((ext_vector_type(16))) __bf16      v16bf;
typedef __attribute__((ext_vector_type(8)))  float       v8f;
typedef __attribute__((ext_vector_type(4)))  unsigned int u32x4;

// Build a 16-element bf16 WMMA fragment from two 16-byte chunks.
__device__ __forceinline__ v16bf frag_from2(const __bf16* p0, const __bf16* p1) {
    u32x4 lo = *(const u32x4*)p0;
    u32x4 hi = *(const u32x4*)p1;
    auto t = __builtin_shufflevector(lo, hi, 0, 1, 2, 3, 4, 5, 6, 7);
    return __builtin_bit_cast(v16bf, t);
}

// ---------------------------------------------------------------------------
// 3-NN (brute force) with inverse-distance weights, one thread per target pt.
// ---------------------------------------------------------------------------
__global__ void knn3_kernel(const float* __restrict__ tgt,   // (16,N,3)
                            const float* __restrict__ src,   // (16,M,3)
                            int N, int M,
                            int* __restrict__ idx, float* __restrict__ wgt)
{
    int id = blockIdx.x * blockDim.x + threadIdx.x;
    if (id >= 16 * N) return;
    int b = id / N;
    float tx = tgt[3 * id + 0], ty = tgt[3 * id + 1], tz = tgt[3 * id + 2];
    const float* s = src + (size_t)b * M * 3;
    float d0 = 3.0e38f, d1 = 3.0e38f, d2 = 3.0e38f;
    int   i0 = 0, i1 = 0, i2 = 0;
    for (int j = 0; j < M; ++j) {
        float dx = s[3 * j + 0] - tx;
        float dy = s[3 * j + 1] - ty;
        float dz = s[3 * j + 2] - tz;
        float d  = dx * dx + dy * dy + dz * dz;
        if (d < d0)      { d2 = d1; i2 = i1; d1 = d0; i1 = i0; d0 = d; i0 = j; }
        else if (d < d1) { d2 = d1; i2 = i1; d1 = d;  i1 = j; }
        else if (d < d2) { d2 = d;  i2 = j; }
    }
    float w0 = 1.0f / fmaxf(sqrtf(d0), 1e-8f);
    float w1 = 1.0f / fmaxf(sqrtf(d1), 1e-8f);
    float w2 = 1.0f / fmaxf(sqrtf(d2), 1e-8f);
    float wsum = w0 + w1 + w2;
    idx[3 * id + 0] = i0; idx[3 * id + 1] = i1; idx[3 * id + 2] = i2;
    wgt[3 * id + 0] = w0 / wsum; wgt[3 * id + 1] = w1 / wsum; wgt[3 * id + 2] = w2 / wsum;
}

// ---------------------------------------------------------------------------
// Weighted gather of 3 source rows + optional self-feature concat,
// output written as bf16 row-major (rows = 16*N, cols = Kout).
// One thread handles 4 channels of one point.  (Used for stages 1 and 2.)
// ---------------------------------------------------------------------------
__global__ void gather_concat_kernel(const float* __restrict__ sfeat,     // (16,M,384)
                                     const float* __restrict__ self_feat, // (16,N,384) or null
                                     const int*   __restrict__ idx,
                                     const float* __restrict__ wgt,
                                     __bf16* __restrict__ Xout,
                                     int N, int M, int Kout, int coff)
{
    int id = blockIdx.x * blockDim.x + threadIdx.x;
    int total = 16 * N * (EMBED / 4);
    if (id >= total) return;
    int p  = id / (EMBED / 4);
    int c0 = (id % (EMBED / 4)) * 4;
    int b  = p / N;
    int   i0 = idx[3 * p + 0], i1 = idx[3 * p + 1], i2 = idx[3 * p + 2];
    float w0 = wgt[3 * p + 0], w1 = wgt[3 * p + 1], w2 = wgt[3 * p + 2];
    const float* base = sfeat + (size_t)b * M * EMBED;
    float4 f0 = *(const float4*)(base + (size_t)i0 * EMBED + c0);
    float4 f1 = *(const float4*)(base + (size_t)i1 * EMBED + c0);
    float4 f2 = *(const float4*)(base + (size_t)i2 * EMBED + c0);
    union { __bf16 h[4]; unsigned long long u; } pk;
    pk.h[0] = (__bf16)(w0 * f0.x + w1 * f1.x + w2 * f2.x);
    pk.h[1] = (__bf16)(w0 * f0.y + w1 * f1.y + w2 * f2.y);
    pk.h[2] = (__bf16)(w0 * f0.z + w1 * f1.z + w2 * f2.z);
    pk.h[3] = (__bf16)(w0 * f0.w + w1 * f1.w + w2 * f2.w);
    *(unsigned long long*)(Xout + (size_t)p * Kout + coff + c0) = pk.u;
    if (self_feat) {
        float4 sf = *(const float4*)(self_feat + (size_t)p * EMBED + c0);
        pk.h[0] = (__bf16)sf.x; pk.h[1] = (__bf16)sf.y;
        pk.h[2] = (__bf16)sf.z; pk.h[3] = (__bf16)sf.w;
        *(unsigned long long*)(Xout + (size_t)p * Kout + c0) = pk.u;
    }
}

__global__ void f32_to_bf16_kernel(const float* __restrict__ a, __bf16* __restrict__ o, int n)
{
    int id = blockIdx.x * blockDim.x + threadIdx.x;
    if (id < n) o[id] = (__bf16)a[id];
}

// ---------------------------------------------------------------------------
// Fused 2-layer MLP: out = relu(X @ W1^T + b1) @ W2^T + b2
// X: (rows, K1) bf16 row-major.  Workgroup = 256 threads (8 waves) owns 64
// rows; hidden tile in LDS.  Wave w -> m-tile (w&3), n-half (w>>2).
// (Used for stages 1 and 2, K1 = 768.)
// ---------------------------------------------------------------------------
template <int K1>
__global__ __launch_bounds__(256)
void mlp2_wmma_kernel(const __bf16* __restrict__ X,
                      const __bf16* __restrict__ W1, const float* __restrict__ B1,
                      const __bf16* __restrict__ W2, const float* __restrict__ B2,
                      float* __restrict__ out)
{
    constexpr int LDH = EMBED + 8;                    // 392 bf16 = 784 B row stride
    __shared__ __attribute__((aligned(16))) __bf16 Hs[64 * LDH];

    const int lane = threadIdx.x & 31;
    const int wave = threadIdx.x >> 5;
    const int mt   = wave & 3;
    const int nh   = wave >> 2;
    const int m    = lane & 15;
    const int half = lane >> 4;
    const size_t rowBase = (size_t)blockIdx.x * 64;

    const __bf16* Xrow = X + (rowBase + mt * 16 + m) * (size_t)K1;

    for (int nt = nh * 12; nt < nh * 12 + 12; ++nt) {
        const int col0 = nt * 16;
        const __bf16* Wrow = W1 + (size_t)(col0 + m) * K1;
        v8f acc = {};
        for (int kt = 0; kt < K1 / 32; ++kt) {
            const __bf16* ap = Xrow + kt * 32 + half * 16;
            v16bf a = frag_from2(ap, ap + 8);
            v16bf bmat = *(const v16bf*)(Wrow + kt * 32 + half * 16);
            acc = __builtin_amdgcn_wmma_f32_16x16x32_bf16(
                      false, a, false, bmat, (short)0, acc, false, false);
        }
        const float bias = B1[col0 + m];
#pragma unroll
        for (int j = 0; j < 8; ++j) {
            float v = acc[j] + bias;
            v = v > 0.0f ? v : 0.0f;
            Hs[(mt * 16 + half * 8 + j) * LDH + col0 + m] = (__bf16)v;
        }
    }
    __syncthreads();

    for (int nt = nh * 12; nt < nh * 12 + 12; ++nt) {
        const int col0 = nt * 16;
        const __bf16* Wrow = W2 + (size_t)(col0 + m) * EMBED;
        v8f acc = {};
        for (int kt = 0; kt < EMBED / 32; ++kt) {
            const __bf16* ap = &Hs[(mt * 16 + m) * LDH + kt * 32 + half * 16];
            v16bf a = frag_from2(ap, ap + 8);
            v16bf bmat = *(const v16bf*)(Wrow + kt * 32 + half * 16);
            acc = __builtin_amdgcn_wmma_f32_16x16x32_bf16(
                      false, a, false, bmat, (short)0, acc, false, false);
        }
        const float bias = B2[col0 + m];
        float* orow = out + (rowBase + mt * 16 + half * 8) * (size_t)EMBED + col0 + m;
#pragma unroll
        for (int j = 0; j < 8; ++j)
            orow[(size_t)j * EMBED] = acc[j] + bias;
    }
}

// ---------------------------------------------------------------------------
// Stage 3 fused: 3-NN interpolation (gather from sfeat with precomputed
// idx/wgt) -> LDS A-tile (bf16) -> 2-layer MLP -> d_out.
// Block = 256 threads (8 waves) owns 32 rows: 2 m-tiles x 24 n-tiles,
// wave w -> m-tile (w&1), n-quarter (w>>1) -> 6 n-tiles.
// Avoids a 100 MB intermediate activation round-trip through HBM.
// ---------------------------------------------------------------------------
__global__ __launch_bounds__(256)
void interp_mlp2_fused_kernel(const float* __restrict__ sfeat,  // (16,M,384) f32
                              const int*   __restrict__ idx,
                              const float* __restrict__ wgt,
                              const __bf16* __restrict__ W1, const float* __restrict__ B1,
                              const __bf16* __restrict__ W2, const float* __restrict__ B2,
                              float* __restrict__ out, int N, int M)
{
    constexpr int LDH = EMBED + 8;                    // 392 bf16 = 784 B row stride
    __shared__ __attribute__((aligned(16))) __bf16 As[32 * LDH];
    __shared__ __attribute__((aligned(16))) __bf16 Hs[32 * LDH];

    const size_t rowBase = (size_t)blockIdx.x * 32;

    // ---- interpolate 32 rows straight into LDS (bf16) ----
    // 32 rows * 96 float4-chunks = 3072 chunks = 12 per thread (exact).
#pragma unroll 1
    for (int chunk = threadIdx.x; chunk < 32 * (EMBED / 4); chunk += 256) {
        const int r  = chunk / (EMBED / 4);
        const int c0 = (chunk % (EMBED / 4)) * 4;
        const size_t p = rowBase + r;
        const int b  = (int)(p / (size_t)N);
        const int   i0 = idx[3 * p + 0], i1 = idx[3 * p + 1], i2 = idx[3 * p + 2];
        const float w0 = wgt[3 * p + 0], w1 = wgt[3 * p + 1], w2 = wgt[3 * p + 2];
        const float* base = sfeat + (size_t)b * M * EMBED;
        float4 f0 = *(const float4*)(base + (size_t)i0 * EMBED + c0);
        float4 f1 = *(const float4*)(base + (size_t)i1 * EMBED + c0);
        float4 f2 = *(const float4*)(base + (size_t)i2 * EMBED + c0);
        union { __bf16 h[4]; unsigned long long u; } pk;
        pk.h[0] = (__bf16)(w0 * f0.x + w1 * f1.x + w2 * f2.x);
        pk.h[1] = (__bf16)(w0 * f0.y + w1 * f1.y + w2 * f2.y);
        pk.h[2] = (__bf16)(w0 * f0.z + w1 * f1.z + w2 * f2.z);
        pk.h[3] = (__bf16)(w0 * f0.w + w1 * f1.w + w2 * f2.w);
        *(unsigned long long*)&As[r * LDH + c0] = pk.u;
    }
    __syncthreads();

    const int lane = threadIdx.x & 31;
    const int wave = threadIdx.x >> 5;
    const int mt   = wave & 1;                        // m-tile 0..1
    const int nq   = wave >> 1;                       // n-quarter 0..3
    const int m    = lane & 15;
    const int half = lane >> 4;

    // ---- Layer 1: H = relu(A @ W1^T + b1) -> LDS ----
    for (int nt = nq * 6; nt < nq * 6 + 6; ++nt) {
        const int col0 = nt * 16;
        const __bf16* Wrow = W1 + (size_t)(col0 + m) * EMBED;
        v8f acc = {};
        for (int kt = 0; kt < EMBED / 32; ++kt) {
            const __bf16* ap = &As[(mt * 16 + m) * LDH + kt * 32 + half * 16];
            v16bf a = frag_from2(ap, ap + 8);
            v16bf bmat = *(const v16bf*)(Wrow + kt * 32 + half * 16);
            acc = __builtin_amdgcn_wmma_f32_16x16x32_bf16(
                      false, a, false, bmat, (short)0, acc, false, false);
        }
        const float bias = B1[col0 + m];
#pragma unroll
        for (int j = 0; j < 8; ++j) {
            float v = acc[j] + bias;
            v = v > 0.0f ? v : 0.0f;
            Hs[(mt * 16 + half * 8 + j) * LDH + col0 + m] = (__bf16)v;
        }
    }
    __syncthreads();

    // ---- Layer 2: out = H @ W2^T + b2 ----
    for (int nt = nq * 6; nt < nq * 6 + 6; ++nt) {
        const int col0 = nt * 16;
        const __bf16* Wrow = W2 + (size_t)(col0 + m) * EMBED;
        v8f acc = {};
        for (int kt = 0; kt < EMBED / 32; ++kt) {
            const __bf16* ap = &Hs[(mt * 16 + m) * LDH + kt * 32 + half * 16];
            v16bf a = frag_from2(ap, ap + 8);
            v16bf bmat = *(const v16bf*)(Wrow + kt * 32 + half * 16);
            acc = __builtin_amdgcn_wmma_f32_16x16x32_bf16(
                      false, a, false, bmat, (short)0, acc, false, false);
        }
        const float bias = B2[col0 + m];
        float* orow = out + (rowBase + mt * 16 + half * 8) * (size_t)EMBED + col0 + m;
#pragma unroll
        for (int j = 0; j < 8; ++j)
            orow[(size_t)j * EMBED] = acc[j] + bias;
    }
}

// ---------------------------------------------------------------------------
extern "C" void kernel_launch(void* const* d_in, const int* in_sizes, int n_in,
                              void* d_out, int out_size, void* d_ws, size_t ws_size,
                              hipStream_t stream)
{
    (void)in_sizes; (void)n_in; (void)out_size; (void)ws_size;
    const float* feat1 = (const float*)d_in[0];
    const float* feat2 = (const float*)d_in[1];
    const float* feat3 = (const float*)d_in[2];
    const float* ctr1  = (const float*)d_in[3];
    const float* ctr2  = (const float*)d_in[4];
    const float* ctr3  = (const float*)d_in[5];
    const float* xyz   = (const float*)d_in[6];
    const float* w32_1 = (const float*)d_in[7];
    const float* b32_1 = (const float*)d_in[8];
    const float* w32_2 = (const float*)d_in[9];
    const float* b32_2 = (const float*)d_in[10];
    const float* w21_1 = (const float*)d_in[11];
    const float* b21_1 = (const float*)d_in[12];
    const float* w21_2 = (const float*)d_in[13];
    const float* b21_2 = (const float*)d_in[14];
    const float* w1N_1 = (const float*)d_in[15];
    const float* b1N_1 = (const float*)d_in[16];
    const float* w1N_2 = (const float*)d_in[17];
    const float* b1N_2 = (const float*)d_in[18];

    char* ws = (char*)d_ws;
    auto carve = [&](size_t bytes) {
        char* p = ws;
        ws += (bytes + 255) & ~(size_t)255;
        return p;
    };
    const size_t SZ_BIG = (size_t)EMBED * 2 * EMBED;  // 294912 elems
    const size_t SZ_SQ  = (size_t)EMBED * EMBED;      // 147456 elems
    __bf16* wb32_1 = (__bf16*)carve(SZ_BIG * 2);
    __bf16* wb32_2 = (__bf16*)carve(SZ_SQ * 2);
    __bf16* wb21_1 = (__bf16*)carve(SZ_BIG * 2);
    __bf16* wb21_2 = (__bf16*)carve(SZ_SQ * 2);
    __bf16* wb1N_1 = (__bf16*)carve(SZ_SQ * 2);
    __bf16* wb1N_2 = (__bf16*)carve(SZ_SQ * 2);
    int*    idxb   = (int*)carve((size_t)3 * 16 * 8192 * sizeof(int));
    float*  wgtb   = (float*)carve((size_t)3 * 16 * 8192 * sizeof(float));
    __bf16* Xbuf   = (__bf16*)carve((size_t)16 * 512 * 2 * EMBED * 2); // stage-1/2 activations
    float*  Fbuf   = (float*)carve((size_t)16 * 512 * EMBED * 4);      // inter-stage f32

    auto cvt = [&](const float* s, __bf16* d, int n) {
        f32_to_bf16_kernel<<<(n + 255) / 256, 256, 0, stream>>>(s, d, n);
    };
    cvt(w32_1, wb32_1, (int)SZ_BIG);
    cvt(w32_2, wb32_2, (int)SZ_SQ);
    cvt(w21_1, wb21_1, (int)SZ_BIG);
    cvt(w21_2, wb21_2, (int)SZ_SQ);
    cvt(w1N_1, wb1N_1, (int)SZ_SQ);
    cvt(w1N_2, wb1N_2, (int)SZ_SQ);

    // Stage 1: level3 -> level2 : interp(ctr2<-ctr3, feat3), concat feat2, MLP32
    {
        const int N = 256, M = 64;
        knn3_kernel<<<(16 * N + 255) / 256, 256, 0, stream>>>(ctr2, ctr3, N, M, idxb, wgtb);
        int tot = 16 * N * (EMBED / 4);
        gather_concat_kernel<<<(tot + 255) / 256, 256, 0, stream>>>(
            feat3, feat2, idxb, wgtb, Xbuf, N, M, 2 * EMBED, EMBED);
        mlp2_wmma_kernel<2 * EMBED><<<(16 * N) / 64, 256, 0, stream>>>(
            Xbuf, wb32_1, b32_1, wb32_2, b32_2, Fbuf);
    }
    // Stage 2: level2 -> level1 : interp(ctr1<-ctr2, fused_2), concat feat1, MLP21
    {
        const int N = 512, M = 256;
        knn3_kernel<<<(16 * N + 255) / 256, 256, 0, stream>>>(ctr1, ctr2, N, M, idxb, wgtb);
        int tot = 16 * N * (EMBED / 4);
        gather_concat_kernel<<<(tot + 255) / 256, 256, 0, stream>>>(
            Fbuf, feat1, idxb, wgtb, Xbuf, N, M, 2 * EMBED, EMBED);
        mlp2_wmma_kernel<2 * EMBED><<<(16 * N) / 64, 256, 0, stream>>>(
            Xbuf, wb21_1, b21_1, wb21_2, b21_2, Fbuf);
    }
    // Stage 3 (fused): interp(xyz<-ctr1, fused_1) -> LDS -> MLP1N -> d_out
    {
        const int N = 8192, M = 512;
        knn3_kernel<<<(16 * N + 255) / 256, 256, 0, stream>>>(xyz, ctr1, N, M, idxb, wgtb);
        interp_mlp2_fused_kernel<<<(16 * N) / 32, 256, 0, stream>>>(
            Fbuf, idxb, wgtb, wb1N_1, b1N_1, wb1N_2, b1N_2, (float*)d_out, N, M);
    }
}
